// FinePreprocess_45612552683783
// MI455X (gfx1250) — compile-verified, matrix operand order checked by reading, not run on hardware
//
#include <hip/hip_runtime.h>

// ---------------------------------------------------------------------------
// FinePreprocess fused kernel for MI455X (gfx1250, wave32, WMMA bf16)
//
//   out[n,p,c] = sum_k  fineWin[n,p,k] * merge_w[c,k]          (k < 128)
//              + sum_k  coarse[n,k]    * W_comb[c,k]           (k < 256)
//              + b_comb[c]
//   W_comb = merge_w[:,128:256] @ proj_w ,  b_comb = merge_w[:,128:]@proj_b + merge_b
// ---------------------------------------------------------------------------

typedef __attribute__((ext_vector_type(8)))  float        v8f;
typedef __attribute__((ext_vector_type(16))) __bf16       v16bf;
typedef __attribute__((ext_vector_type(4)))  unsigned int u32x4;

union Frag {
    u32x4 u[2];
    v16bf v;
};

#define WW        5          // window size
#define CF        128        // fine channels
#define CC        256        // coarse channels
#define HF        416
#define WF        416
#define LDIM      10816      // coarse tokens per image (104*104)
#define PAD_K     144        // LDS halfwords per row (128 data + 16 pad -> no bank conflicts)
#define PAD_KW    (PAD_K/2)  // dwords per row

// --------------------------- prepass kernels -------------------------------

// W_combT[k][c] = sum_j merge_w[c, 128+j] * proj_w[j, k]   (stored bf16, [256][128])
__global__ __launch_bounds__(256) void prep_wcombT(const float* __restrict__ merge_w,
                                                   const float* __restrict__ proj_w,
                                                   __bf16* __restrict__ W_combT) {
    int idx = blockIdx.x * 256 + threadIdx.x;      // 0 .. 32767
    int k = idx >> 7;                              // 0 .. 255
    int c = idx & 127;                             // 0 .. 127
    float acc = 0.f;
    #pragma unroll 4
    for (int j = 0; j < CF; ++j)
        acc = fmaf(merge_w[c * CC + CF + j], proj_w[j * CC + k], acc);
    W_combT[k * CF + c] = (__bf16)acc;
}

// A_w (bf16 copy of merge_w[:,0:128]) and b_comb
__global__ __launch_bounds__(256) void prep_misc(const float* __restrict__ merge_w,
                                                 const float* __restrict__ merge_b,
                                                 const float* __restrict__ proj_b,
                                                 float* __restrict__ b_comb,
                                                 __bf16* __restrict__ A_w) {
    int tid = blockIdx.x * 256 + threadIdx.x;      // 0 .. 16383
    int c = tid >> 7;
    int k = tid & 127;
    A_w[c * CF + k] = (__bf16)merge_w[c * CC + k];
    if (tid < CF) {
        float acc = merge_b[tid];
        #pragma unroll 4
        for (int j = 0; j < CF; ++j)
            acc = fmaf(merge_w[tid * CC + CF + j], proj_b[j], acc);
        b_comb[tid] = acc;
    }
}

// ------------------------------ main kernel --------------------------------

__global__ __launch_bounds__(256) void fine_preprocess_kernel(
    const float* __restrict__ coarse0, const float* __restrict__ coarse1,
    const float* __restrict__ fine0,   const float* __restrict__ fine1,
    const int* __restrict__ b_idx, const int* __restrict__ i_idx,
    const int* __restrict__ j_idx, const int* __restrict__ stride_p,
    const __bf16* __restrict__ W_combT, const float* __restrict__ b_comb,
    const __bf16* __restrict__ A_w, float* __restrict__ out, int N) {

    __shared__ __bf16 sA[32 * PAD_K];    // gathered window (rows p=0..31, bf16)  9216 B
    __shared__ __bf16 sB[128 * PAD_K];   // merge weights, row c = out channel   36864 B
    __shared__ float  sCoarse[CC];
    __shared__ float  sPart[256];
    __shared__ float  sD[CF];

    const int tid = threadIdx.x;
    const int n   = blockIdx.x;          // 0 .. 2N-1
    const int sel = (n >= N);
    const int m   = sel ? (n - N) : n;

    const int b    = b_idx[m];
    const int fidx = sel ? j_idx[m] : i_idx[m];
    const int str  = stride_p[0];
    const int pad  = WW / 2;
    const int Wc   = (WF + 2 * pad - WW) / str + 1;

    const float* __restrict__ coarse = (sel ? coarse1 : coarse0)
                                     + ((size_t)b * LDIM + fidx) * CC;
    const float* __restrict__ fb = (sel ? fine1 : fine0) + (size_t)b * CF * HF * WF;

    // --- stage merge weights into LDS (padded rows) ------------------------
    {
        const unsigned int* src = (const unsigned int*)A_w;   // 8192 dwords
        unsigned int* dst = (unsigned int*)sB;
        #pragma unroll
        for (int it = 0; it < 32; ++it) {
            int i = tid + it * 256;
            int c  = i >> 6;          // 64 dwords per source row
            int kw = i & 63;
            dst[c * PAD_KW + kw] = src[i];
        }
    }

    // --- coarse vector -----------------------------------------------------
    sCoarse[tid] = coarse[tid];

    // --- zero pad rows 25..31 of the window matrix -------------------------
    for (int i = tid; i < 7 * 64; i += 256) {
        int r = 25 + (i >> 6);
        ((unsigned int*)sA)[r * PAD_KW + (i & 63)] = 0u;
    }

    // --- gather 5x5x128 window -> LDS bf16 ---------------------------------
    {
        const int row = fidx / Wc, col = fidx % Wc;
        const int y0 = row * str - pad, x0 = col * str - pad;
        for (int task = tid; task < CF * WW; task += 256) {   // 640 tasks: (c, dy)
            int c  = task & (CF - 1);
            int dy = task >> 7;
            int y  = y0 + dy;
            bool yok = (y >= 0) & (y < HF);
            const float* rowp = fb + ((size_t)c * HF + y) * WF;
            #pragma unroll
            for (int dx = 0; dx < WW; ++dx) {
                int x = x0 + dx;
                float v = 0.f;
                if (yok && x >= 0 && x < WF) v = rowp[x];
                sA[(dy * WW + dx) * PAD_K + c] = (__bf16)v;
            }
        }
    }
    __syncthreads();

    // --- shared coarse contribution: d[c] = b_comb[c] + W_comb[c,:]·coarse --
    {
        int c = tid & 127, half = tid >> 7;
        float acc = 0.f;
        const __bf16* wt = W_combT + (size_t)half * 128 * CF + c;
        #pragma unroll 4
        for (int kk = 0; kk < 128; ++kk)
            acc = fmaf((float)wt[kk * CF], sCoarse[half * 128 + kk], acc);
        sPart[tid] = acc;
    }
    __syncthreads();
    if (tid < CF) sD[tid] = sPart[tid] + sPart[tid + 128] + b_comb[tid];
    __syncthreads();

    // --- WMMA: [32 x 128] = sA[32 x 128] @ sB^T, bf16 in / f32 acc ---------
    const int wave = tid >> 5;           // 0..7 -> 16-column block
    const int lane = tid & 31;
    const int g    = lane >> 4;          // lane group (K-halves per ISA layout)
    const int mrow = lane & 15;
    const int c0   = wave * 16;

    const __bf16* arow0 = sA + mrow * PAD_K;            // row tile p = 0..15
    const __bf16* arow1 = sA + (16 + mrow) * PAD_K;     // row tile p = 16..31
    const __bf16* brow  = sB + (c0 + mrow) * PAD_K;     // B column = out channel

    v8f acc0 = {}, acc1 = {};
    #pragma unroll
    for (int kb = 0; kb < 4; ++kb) {
        const int k0 = kb * 32;
        Frag a0, a1, bb;
        a0.u[0] = *(const u32x4*)(arow0 + k0 + 8 * g);
        a0.u[1] = *(const u32x4*)(arow0 + k0 + 16 + 8 * g);
        a1.u[0] = *(const u32x4*)(arow1 + k0 + 8 * g);
        a1.u[1] = *(const u32x4*)(arow1 + k0 + 16 + 8 * g);
        bb.u[0] = *(const u32x4*)(brow  + k0 + 8 * g);
        bb.u[1] = *(const u32x4*)(brow  + k0 + 16 + 8 * g);
        acc0 = __builtin_amdgcn_wmma_f32_16x16x32_bf16(false, a0.v, false, bb.v,
                                                       (short)0, acc0, false, false);
        acc1 = __builtin_amdgcn_wmma_f32_16x16x32_bf16(false, a1.v, false, bb.v,
                                                       (short)0, acc1, false, false);
    }

    // --- writeback: out[n, p, c] = acc + d[c]  (coalesced over c) ----------
    const float dv = sD[c0 + mrow];
    float* op = out + (size_t)n * (WW * WW * CF);
    #pragma unroll
    for (int r = 0; r < 8; ++r) {
        int p0 = r + 8 * g;                         // 0..15
        op[p0 * CF + c0 + mrow] = acc0[r] + dv;
        int p1 = 16 + p0;                           // 16..31, clip to 25
        if (p1 < WW * WW) op[p1 * CF + c0 + mrow] = acc1[r] + dv;
    }
}

// ------------------------------- launcher ----------------------------------

extern "C" void kernel_launch(void* const* d_in, const int* in_sizes, int n_in,
                              void* d_out, int out_size, void* d_ws, size_t ws_size,
                              hipStream_t stream) {
    const float* coarse0 = (const float*)d_in[0];
    const float* coarse1 = (const float*)d_in[1];
    const float* fine0   = (const float*)d_in[2];
    const float* fine1   = (const float*)d_in[3];
    const int*   b_idx   = (const int*)d_in[4];
    const int*   i_idx   = (const int*)d_in[5];
    const int*   j_idx   = (const int*)d_in[6];
    const int*   strd    = (const int*)d_in[7];
    const float* proj_w  = (const float*)d_in[8];
    const float* proj_b  = (const float*)d_in[9];
    const float* merge_w = (const float*)d_in[10];
    const float* merge_b = (const float*)d_in[11];
    float*       out     = (float*)d_out;

    const int N = in_sizes[4];

    // workspace layout
    __bf16* W_combT = (__bf16*)d_ws;                              // 256*128*2 = 65536 B
    float*  b_comb  = (float*)((char*)d_ws + 65536);              // 512 B
    __bf16* A_w     = (__bf16*)((char*)d_ws + 65536 + 512);       // 128*128*2 = 32768 B

    prep_wcombT<<<dim3(128), dim3(256), 0, stream>>>(merge_w, proj_w, W_combT);
    prep_misc  <<<dim3(64),  dim3(256), 0, stream>>>(merge_w, merge_b, proj_b, b_comb, A_w);

    fine_preprocess_kernel<<<dim3(2 * N), dim3(256), 0, stream>>>(
        coarse0, coarse1, fine0, fine1, b_idx, i_idx, j_idx, strd,
        W_combT, b_comb, A_w, out, N);
}
